// InvariantPointAttention_78228534329677
// MI455X (gfx1250) — matrix operand order repaired and checked by reading, not compiled, not run on hardware
//
#include <hip/hip_runtime.h>
#include <hip/hip_bf16.h>

#define DEVFN __device__ __forceinline__

typedef __attribute__((ext_vector_type(16))) _Float16 v16h;
typedef __attribute__((ext_vector_type(8)))  float    v8f;
typedef _Float16 f16;

#if defined(__has_builtin)
#if __has_builtin(__builtin_amdgcn_global_load_async_to_lds_b128) && \
    __has_builtin(__builtin_amdgcn_s_wait_asynccnt)
#define USE_ASYNC_LDS 1
#endif
#endif
#ifndef USE_ASYNC_LDS
#define USE_ASYNC_LDS 0
#endif

constexpr int kB   = 2;
constexpr int kL   = 512;
constexpr int kCS  = 384;
constexpr int kCZ  = 128;
constexpr int kH   = 8;
constexpr int kPQK = 4;
constexpr int kPV  = 8;
constexpr int kHD  = kCS / kH;                   // 48
constexpr int kOUT = kCS + kH * kPV * 3 + kH * kCZ; // 1600
constexpr int kM   = kB * kL;                    // 1024
constexpr int kEK  = 64;                         // extended QK depth: 48 + 12 + 4 pad
constexpr int kVC  = 80;                         // concat V width: 48 + 24 + 8 pad

// ---------------- WMMA helpers (CDNA5 f16 16x16x32, f32 accum) ----------------
DEVFN v8f wmma_f16(v16h a, v16h b, v8f c) {
  return __builtin_amdgcn_wmma_f32_16x16x32_f16(false, a, false, b, (short)0, c,
                                                false, false);
}
// A-matrix 16x32 f16: lane l holds row (l&15); element t -> K index:
DEVFN int ka_idx(int t, int kh) { return ((t & 8) << 1) + kh * 8 + (t & 7); }
// B-matrix 32x16 f16: lane l holds col (l&15); element t -> K index:
DEVFN int kb_idx(int t, int kh) { return ((t & 8) << 1) + kh * 8 + (t & 7); }

#if USE_ASYNC_LDS
typedef int v4i_ __attribute__((vector_size(16)));
DEVFN void async_copy16(const float* g, float* lds) {
  __builtin_amdgcn_global_load_async_to_lds_b128(
      (__attribute__((address_space(1))) v4i_*)(unsigned long long)g,
      (__attribute__((address_space(3))) v4i_*)(unsigned)(unsigned long long)lds,
      0, 0);
}
#define ASYNC_WAIT(n) __builtin_amdgcn_s_wait_asynccnt(n)
#else
#define ASYNC_WAIT(n)
#endif

// ---------------- 1) LayerNorm ----------------
__global__ __launch_bounds__(128)
void ln_kernel(const float* __restrict__ x, const float* __restrict__ g,
               const float* __restrict__ b, float* __restrict__ y) {
  int row = blockIdx.x, t = threadIdx.x;
  const float* xr = x + (size_t)row * kCS;
  float v0 = xr[t], v1 = xr[t + 128], v2 = xr[t + 256];
  __shared__ float red[128];
  red[t] = v0 + v1 + v2;
  __syncthreads();
  for (int s = 64; s; s >>= 1) { if (t < s) red[t] += red[t + s]; __syncthreads(); }
  float mu = red[0] * (1.0f / kCS);
  __syncthreads();
  float d0 = v0 - mu, d1 = v1 - mu, d2 = v2 - mu;
  red[t] = d0 * d0 + d1 * d1 + d2 * d2;
  __syncthreads();
  for (int s = 64; s; s >>= 1) { if (t < s) red[t] += red[t + s]; __syncthreads(); }
  float inv = rsqrtf(red[0] * (1.0f / kCS) + 1e-5f);
  float* yr = y + (size_t)row * kCS;
  yr[t]       = d0 * inv * g[t]       + b[t];
  yr[t + 128] = d1 * inv * g[t + 128] + b[t + 128];
  yr[t + 256] = d2 * inv * g[t + 256] + b[t + 256];
}

// ---------------- 2) generic WMMA GEMM: C[M,N] = A[M,K] @ W[K,N] + bias -------
__global__ __launch_bounds__(32)
void gemm_rrr_f16(const float* __restrict__ A, const float* __restrict__ Bm,
                  const float* __restrict__ bias, float* __restrict__ C,
                  int N, int K) {
  int ntn = N >> 4;
  int ti = blockIdx.x / ntn, tj = blockIdx.x % ntn;
  int l = threadIdx.x, lm = l & 15, kh = l >> 4;
  const float* arow = A + (size_t)(ti * 16 + lm) * K;
  const float* bcol = Bm + (tj * 16 + lm);
  v8f acc = {};
  for (int k0 = 0; k0 < K; k0 += 32) {
    v16h af, bf;
#pragma unroll
    for (int t = 0; t < 16; ++t) af[t] = (f16)arow[k0 + ka_idx(t, kh)];
#pragma unroll
    for (int t = 0; t < 16; ++t) bf[t] = (f16)bcol[(size_t)(k0 + kb_idx(t, kh)) * N];
    acc = wmma_f16(af, bf, acc);
  }
  int col = tj * 16 + lm;
  float bn = bias ? bias[col] : 0.0f;
#pragma unroll
  for (int r = 0; r < 8; ++r)
    C[(size_t)(ti * 16 + kh * 8 + r) * N + col] = acc[r] + bn;
}

// ---------------- 3) apply frames: out = R p + t ----------------
__global__ __launch_bounds__(256)
void frames_kernel(const float* __restrict__ pts, const float* __restrict__ rots,
                   const float* __restrict__ trans, float* __restrict__ out,
                   int n, int HP) {
  int idx = blockIdx.x * blockDim.x + threadIdx.x;
  if (idx >= n) return;
  int bl = idx / HP;
  const float* R = rots + (size_t)bl * 9;
  const float* t = trans + (size_t)bl * 3;
  const float* p = pts + (size_t)idx * 3;
  float x = p[0], y = p[1], z = p[2];
  out[(size_t)idx * 3 + 0] = R[0] * x + R[1] * y + R[2] * z + t[0];
  out[(size_t)idx * 3 + 1] = R[3] * x + R[4] * y + R[5] * z + t[1];
  out[(size_t)idx * 3 + 2] = R[6] * x + R[7] * y + R[8] * z + t[2];
}

// ---------------- 4) build Qe/Ke (f16, K=64), Vc (f16, 80), qn/kn ------------
__global__ __launch_bounds__(256)
void prep_kernel(const float* __restrict__ q, const float* __restrict__ k,
                 const float* __restrict__ v, const float* __restrict__ qg,
                 const float* __restrict__ kg, const float* __restrict__ vg,
                 const float* __restrict__ hw, f16* __restrict__ qe,
                 f16* __restrict__ ke, f16* __restrict__ vc,
                 float* __restrict__ qn, float* __restrict__ kn) {
  int idx = blockIdx.x * blockDim.x + threadIdx.x;  // over B*H*L
  if (idx >= kB * kH * kL) return;
  int i = idx % kL, bh = idx / kL, h = bh & (kH - 1), b = bh / kH;
  float wc = log1pf(expf(hw[h]));
  float sq = sqrtf(wc);
  const float rs = rsqrtf((float)kHD);
  const float* qrow = q + (size_t)(b * kL + i) * kCS + h * kHD;
  const float* krow = k + (size_t)(b * kL + i) * kCS + h * kHD;
  const float* vrow = v + (size_t)(b * kL + i) * kCS + h * kHD;
  f16* qer = qe + (size_t)idx * kEK;
  f16* ker = ke + (size_t)idx * kEK;
  f16* vcr = vc + (size_t)idx * kVC;
  for (int d = 0; d < kHD; ++d) { qer[d] = (f16)(qrow[d] * rs); ker[d] = (f16)krow[d]; vcr[d] = (f16)vrow[d]; }
  const float* qgp = qg + (size_t)((b * kL + i) * kH + h) * kPQK * 3;
  const float* kgp = kg + (size_t)((b * kL + i) * kH + h) * kPQK * 3;
  float sq2 = 0.f, sk2 = 0.f;
  for (int t = 0; t < kPQK * 3; ++t) {
    float xq = qgp[t], xk = kgp[t];
    sq2 += xq * xq; sk2 += xk * xk;
    qer[48 + t] = (f16)(sq * xq); ker[48 + t] = (f16)(sq * xk);
  }
  for (int t = 60; t < kEK; ++t) { qer[t] = (f16)0.f; ker[t] = (f16)0.f; }
  qn[idx] = sq2; kn[idx] = sk2;
  const float* vgp = vg + (size_t)((b * kL + i) * kH + h) * kPV * 3;
  for (int t = 0; t < kPV * 3; ++t) vcr[48 + t] = (f16)vgp[t];
  for (int t = 72; t < kVC; ++t) vcr[t] = (f16)0.f;
}

// ---------------- 5) pair bias -> attn (f32, [B,H,L,L]) ----------------
__global__ __launch_bounds__(256)
void pair_bias_kernel(const float* __restrict__ pair, const float* __restrict__ Wpb,
                      float* __restrict__ attn) {
  __shared__ float wpb_s[kCZ * kH];
  for (int t = threadIdx.x; t < kCZ * kH; t += blockDim.x) wpb_s[t] = Wpb[t];
  __syncthreads();
  int idx = blockIdx.x * blockDim.x + threadIdx.x;  // over B*L*L
  if (idx >= kB * kL * kL) return;
  int j = idx % kL, bi = idx / kL, i = bi % kL, b = bi / kL;
  const float* prow = pair + (size_t)idx * kCZ;
  float acc[kH] = {};
  for (int c = 0; c < kCZ; ++c) {
    float pv = prow[c];
#pragma unroll
    for (int h = 0; h < kH; ++h) acc[h] += pv * wpb_s[c * kH + h];
  }
#pragma unroll
  for (int h = 0; h < kH; ++h)
    attn[(((size_t)(b * kH + h)) * kL + i) * kL + j] = acc[h];
}

// ---------------- 6) logits: attn += Qe·Ke - 0.5 wc (qn+kn)  (WMMA) ----------
__global__ __launch_bounds__(32)
void logits_kernel(const f16* __restrict__ qe, const f16* __restrict__ ke,
                   const float* __restrict__ qn, const float* __restrict__ kn,
                   const float* __restrict__ hw, float* __restrict__ attn) {
  int blk = blockIdx.x;               // B*H * 32 * 32 tiles
  int tj = blk & 31, ti = (blk >> 5) & 31, bh = blk >> 10;
  int h = bh & (kH - 1);
  int l = threadIdx.x, lm = l & 15, kh = l >> 4;
  const f16* qrow = qe + ((size_t)bh * kL + ti * 16 + lm) * kEK;
  const f16* krow = ke + ((size_t)bh * kL + tj * 16 + lm) * kEK;
  v8f acc = {};
  for (int k0 = 0; k0 < kEK; k0 += 32) {
    v16h af, bf;
#pragma unroll
    for (int t = 0; t < 16; ++t) af[t] = qrow[k0 + ka_idx(t, kh)];
#pragma unroll
    for (int t = 0; t < 16; ++t) bf[t] = krow[k0 + kb_idx(t, kh)];
    acc = wmma_f16(af, bf, acc);
  }
  float wc = log1pf(expf(hw[h]));
  float knj = kn[(size_t)bh * kL + tj * 16 + lm];
#pragma unroll
  for (int r = 0; r < 8; ++r) {
    int i = ti * 16 + kh * 8 + r;
    size_t o = ((size_t)bh * kL + i) * kL + tj * 16 + lm;
    attn[o] += acc[r] - 0.5f * wc * (qn[(size_t)bh * kL + i] + knj);
  }
}

// ---------------- 7) softmax over j; write f16 attention ----------------
__global__ __launch_bounds__(256)
void softmax_kernel(const float* __restrict__ logits, f16* __restrict__ out) {
  int w = threadIdx.x >> 5, lane = threadIdx.x & 31;
  size_t row = (size_t)blockIdx.x * 8 + w;  // B*H*L rows
  const float* p = logits + row * kL;
  float vals[16], mx = -3.0e38f;
#pragma unroll
  for (int t = 0; t < 16; ++t) { vals[t] = p[lane + t * 32]; mx = fmaxf(mx, vals[t]); }
#pragma unroll
  for (int s = 16; s; s >>= 1) mx = fmaxf(mx, __shfl_xor(mx, s));
  float sum = 0.f;
#pragma unroll
  for (int t = 0; t < 16; ++t) { vals[t] = expf(vals[t] - mx); sum += vals[t]; }
#pragma unroll
  for (int s = 16; s; s >>= 1) sum += __shfl_xor(sum, s);
  float inv = 1.0f / sum;
  f16* o = out + row * kL;
#pragma unroll
  for (int t = 0; t < 16; ++t) o[lane + t * 32] = (f16)(vals[t] * inv);
}

// ---------------- 8) oc[b,h,i,80] = attn @ [v ; v_g]  (WMMA) ----------------
__global__ __launch_bounds__(32)
void attn_vc_kernel(const f16* __restrict__ attn16, const f16* __restrict__ vc,
                    float* __restrict__ oc) {
  int blk = blockIdx.x;                 // B*H * 32 * 5
  int tj = blk % 5, ti = (blk / 5) & 31, bh = blk / (5 * 32);
  int l = threadIdx.x, lm = l & 15, kh = l >> 4;
  const f16* arow = attn16 + ((size_t)bh * kL + ti * 16 + lm) * kL;
  const f16* bcol = vc + (size_t)bh * kL * kVC + tj * 16 + lm;
  v8f acc = {};
  for (int k0 = 0; k0 < kL; k0 += 32) {
    v16h af, bf;
#pragma unroll
    for (int t = 0; t < 16; ++t) af[t] = arow[k0 + ka_idx(t, kh)];
#pragma unroll
    for (int t = 0; t < 16; ++t) bf[t] = bcol[(size_t)(k0 + kb_idx(t, kh)) * kVC];
    acc = wmma_f16(af, bf, acc);
  }
#pragma unroll
  for (int r = 0; r < 8; ++r)
    oc[((size_t)bh * kL + ti * 16 + kh * 8 + r) * kVC + tj * 16 + lm] = acc[r];
}

// ------- 9) out_pair: cat[b,i,576+h*128+c] = sum_j attn[b,h,i,j] pair[b,i,j,c]
//         pair is staged into LDS with async global->LDS B128 copies,
//         double-buffered, per-wave slices (no cross-wave sync needed).
__global__ __launch_bounds__(256)
void attn_pair_kernel(const f16* __restrict__ attn16, const float* __restrict__ pair,
                      float* __restrict__ cat) {
  __shared__ float pbuf[2][32 * kCZ];   // 2 x 16KB chunks of pair rows
  int bi = blockIdx.x, b = bi / kL, i = bi % kL;
  int wv = threadIdx.x >> 5;            // n-tile 0..7 over CZ=128
  int l = threadIdx.x & 31, lm = l & 15, kh = l >> 4;
  // A rows: m = h (rows 8..15 padded; we read duplicates and discard outputs)
  const f16* arow = attn16 + (((size_t)(b * kH + (lm & 7))) * kL + i) * kL;
  const float* pbase = pair + (((size_t)b * kL + i) * kL) * kCZ;

  // Each wave stages only the 16-column slice it consumes: 4 B128 ops / chunk.
  auto issue = [&](int k0, int buf) {
#pragma unroll
    for (int s = 0; s < 4; ++s) {
      int idx = s * 32 + l;             // 0..127
      int row = idx >> 2, cs = idx & 3;
      int col = wv * 16 + cs * 4;
      const float* g = pbase + (size_t)(k0 + row) * kCZ + col;
      float* sh = &pbuf[buf][row * kCZ + col];
#if USE_ASYNC_LDS
      async_copy16(g, sh);
#else
      *(float4*)sh = *(const float4*)g;
#endif
    }
  };

  issue(0, 0);
  issue(32, 1);
  v8f acc = {};
  for (int k0 = 0; k0 < kL; k0 += 32) {
    int buf = (k0 >> 5) & 1;
    if (k0 + 32 < kL) { ASYNC_WAIT(4); } else { ASYNC_WAIT(0); }
    v16h af, bf;
#pragma unroll
    for (int t = 0; t < 16; ++t) af[t] = arow[k0 + ka_idx(t, kh)];
#pragma unroll
    for (int t = 0; t < 16; ++t)
      bf[t] = (f16)pbuf[buf][kb_idx(t, kh) * kCZ + wv * 16 + lm];
    acc = wmma_f16(af, bf, acc);
    if (k0 + 64 < kL) issue(k0 + 64, buf);
  }
  if (kh == 0) {                        // rows m=0..7 are the valid heads
#pragma unroll
    for (int r = 0; r < 8; ++r)
      cat[(size_t)bi * kOUT + kCS + kH * kPV * 3 + r * kCZ + wv * 16 + lm] = acc[r];
  }
}

// ------- 10) out_scalar reorder + inverse-frame points into cat --------------
__global__ __launch_bounds__(448)
void finalize_kernel(const float* __restrict__ oc, const float* __restrict__ rots,
                     const float* __restrict__ trans, float* __restrict__ cat) {
  int bl = blockIdx.x, t = threadIdx.x;
  int b = bl / kL, i = bl % kL;
  float* crow = cat + (size_t)bl * kOUT;
  if (t < kCS) {
    int h = t / kHD, d = t % kHD;
    crow[t] = oc[((size_t)(b * kH + h) * kL + i) * kVC + d];
  } else if (t < kCS + kH * kPV) {
    int idx = t - kCS, h = idx / kPV, p = idx % kPV;
    const float* base = oc + ((size_t)(b * kH + h) * kL + i) * kVC + 48 + p * 3;
    const float* R = rots + (size_t)bl * 9;
    const float* tr = trans + (size_t)bl * 3;
    float x = base[0] - tr[0], y = base[1] - tr[1], z = base[2] - tr[2];
    float* o = crow + kCS + h * (kPV * 3) + p * 3;
    o[0] = R[0] * x + R[3] * y + R[6] * z;   // R^T (inverse frame rotation)
    o[1] = R[1] * x + R[4] * y + R[7] * z;
    o[2] = R[2] * x + R[5] * y + R[8] * z;
  }
}

// ======================= launch =======================
extern "C" void kernel_launch(void* const* d_in, const int* in_sizes, int n_in,
                              void* d_out, int out_size, void* d_ws, size_t ws_size,
                              hipStream_t stream) {
  (void)in_sizes; (void)n_in; (void)out_size; (void)ws_size;
  const float* single = (const float*)d_in[0];
  const float* pair   = (const float*)d_in[1];
  const float* rots   = (const float*)d_in[2];
  const float* trans  = (const float*)d_in[3];
  const float* ln_g   = (const float*)d_in[4];
  const float* ln_b   = (const float*)d_in[5];
  const float* Wq     = (const float*)d_in[6];
  const float* Wk     = (const float*)d_in[7];
  const float* Wv     = (const float*)d_in[8];
  const float* Wqp    = (const float*)d_in[9];
  const float* bqp    = (const float*)d_in[10];
  const float* Wkp    = (const float*)d_in[11];
  const float* bkp    = (const float*)d_in[12];
  const float* Wvp    = (const float*)d_in[13];
  const float* bvp    = (const float*)d_in[14];
  const float* Wpb    = (const float*)d_in[15];
  const float* hw     = (const float*)d_in[16];
  const float* Wo     = (const float*)d_in[17];
  const float* bo     = (const float*)d_in[18];
  float* out = (float*)d_out;

  // workspace carve-up (float units)
  float* ws = (float*)d_ws;
  size_t off = 0;
  auto alloc = [&](size_t n) { float* p = ws + off; off += n; return p; };
  float* s_ln = alloc((size_t)kM * kCS);
  float* q    = alloc((size_t)kM * kCS);
  float* k    = alloc((size_t)kM * kCS);
  float* v    = alloc((size_t)kM * kCS);
  float* qp_l = alloc((size_t)kM * kH * kPQK * 3);
  float* kp_l = alloc((size_t)kM * kH * kPQK * 3);
  float* vp_l = alloc((size_t)kM * kH * kPV * 3);
  float* q_g  = alloc((size_t)kM * kH * kPQK * 3);
  float* k_g  = alloc((size_t)kM * kH * kPQK * 3);
  float* v_g  = alloc((size_t)kM * kH * kPV * 3);
  float* attn = alloc((size_t)kB * kH * kL * kL);
  f16* attn16 = (f16*)alloc((size_t)kB * kH * kL * kL / 2);
  f16* qe     = (f16*)alloc((size_t)kB * kH * kL * kEK / 2);
  f16* ke     = (f16*)alloc((size_t)kB * kH * kL * kEK / 2);
  float* qn   = alloc((size_t)kB * kH * kL);
  float* kn   = alloc((size_t)kB * kH * kL);
  f16* vc     = (f16*)alloc((size_t)kB * kH * kL * kVC / 2);
  float* oc   = alloc((size_t)kB * kH * kL * kVC);
  float* cat  = alloc((size_t)kM * kOUT);

  // 1) layer norm
  ln_kernel<<<kM, 128, 0, stream>>>(single, ln_g, ln_b, s_ln);
  // 2) projections (WMMA)
  gemm_rrr_f16<<<(kM / 16) * (kCS / 16), 32, 0, stream>>>(s_ln, Wq, nullptr, q, kCS, kCS);
  gemm_rrr_f16<<<(kM / 16) * (kCS / 16), 32, 0, stream>>>(s_ln, Wk, nullptr, k, kCS, kCS);
  gemm_rrr_f16<<<(kM / 16) * (kCS / 16), 32, 0, stream>>>(s_ln, Wv, nullptr, v, kCS, kCS);
  gemm_rrr_f16<<<(kM / 16) * (96 / 16), 32, 0, stream>>>(s_ln, Wqp, bqp, qp_l, 96, kCS);
  gemm_rrr_f16<<<(kM / 16) * (96 / 16), 32, 0, stream>>>(s_ln, Wkp, bkp, kp_l, 96, kCS);
  gemm_rrr_f16<<<(kM / 16) * (192 / 16), 32, 0, stream>>>(s_ln, Wvp, bvp, vp_l, 192, kCS);
  // 3) frames
  int nq = kM * kH * kPQK, nv = kM * kH * kPV;
  frames_kernel<<<(nq + 255) / 256, 256, 0, stream>>>(qp_l, rots, trans, q_g, nq, kH * kPQK);
  frames_kernel<<<(nq + 255) / 256, 256, 0, stream>>>(kp_l, rots, trans, k_g, nq, kH * kPQK);
  frames_kernel<<<(nv + 255) / 256, 256, 0, stream>>>(vp_l, rots, trans, v_g, nv, kH * kPV);
  // 4) build Qe/Ke/Vc
  prep_kernel<<<(kB * kH * kL) / 256, 256, 0, stream>>>(q, k, v, q_g, k_g, v_g, hw,
                                                        qe, ke, vc, qn, kn);
  // 5) pair bias
  pair_bias_kernel<<<(kB * kL * kL) / 256, 256, 0, stream>>>(pair, Wpb, attn);
  // 6) logits (WMMA)
  logits_kernel<<<kB * kH * 32 * 32, 32, 0, stream>>>(qe, ke, qn, kn, hw, attn);
  // 7) softmax
  softmax_kernel<<<(kB * kH * kL) / 8, 256, 0, stream>>>(attn, attn16);
  // 8) attn @ [v ; v_g] (WMMA)
  attn_vc_kernel<<<kB * kH * 32 * 5, 32, 0, stream>>>(attn16, vc, oc);
  // 9) attn @ pair (WMMA + async global->LDS double buffering)
  attn_pair_kernel<<<kB * kL, 256, 0, stream>>>(attn16, pair, cat);
  // 10) scalar reorder + inverse frames
  finalize_kernel<<<kB * kL, 448, 0, stream>>>(oc, rots, trans, cat);
  // 11) final projection (WMMA, K=1600)
  gemm_rrr_f16<<<(kM / 16) * (kCS / 16), 32, 0, stream>>>(cat, Wo, bo, out, kCS, kOUT);
}
